// NIN_82764019794273
// MI455X (gfx1250) — 8-bit matrix operand order repaired and checked by reading, compiled, not run on hardware
//
#include <hip/hip_runtime.h>
#include <stdint.h>

typedef __attribute__((ext_vector_type(8))) int v8i;

#define BATCH 512
#ifndef USE_ASYNC_LDS
#define USE_ASYNC_LDS 1
#endif

__global__ void conv0_kernel(const float* __restrict__ x, const float* __restrict__ w,
                             const float* __restrict__ bias, float* __restrict__ out) {
  __shared__ float patch[75];
  const int pixel = blockIdx.x;
  const int co    = threadIdx.x;
  const int ow = pixel & 31, oh = (pixel >> 5) & 31, n = pixel >> 10;
  if (co < 75) {
    const int ci = co / 25, rr = co % 25, kh = rr / 5, kw = rr % 5;
    const int ih = oh + kh - 2, iw = ow + kw - 2;
    patch[co] = (ih >= 0 && ih < 32 && iw >= 0 && iw < 32)
                    ? x[((n * 3 + ci) * 32 + ih) * 32 + iw] : 0.f;
  }
  __syncthreads();
  float s = bias[co];
  #pragma unroll
  for (int ci = 0; ci < 3; ++ci) {
    #pragma unroll
    for (int k = 0; k < 25; ++k)
      s += patch[ci * 25 + k] * w[(co * 3 + ci) * 25 + k];
  }
  out[(size_t)pixel * 192 + co] = s;
}

__global__ void bn_part_kernel(const float* __restrict__ in, float* __restrict__ ps,
                               float* __restrict__ pq, int P, int C, int chunk, int nblk) {
  const int c = threadIdx.x;
  const int b = blockIdx.x;
  const int p0 = b * chunk;
  const int p1 = (p0 + chunk) < P ? (p0 + chunk) : P;
  float s = 0.f, q = 0.f;
  for (int p = p0; p < p1; ++p) {
    const float v = in[(size_t)p * C + c];
    s += v;
    q += v * v;
  }
  ps[(size_t)c * nblk + b] = s;
  pq[(size_t)c * nblk + b] = q;
}

__global__ void bn_finish_kernel(const float* __restrict__ ps, const float* __restrict__ pq,
                                 const float* __restrict__ g, const float* __restrict__ be,
                                 float* __restrict__ scale, float* __restrict__ shift,
                                 int P, int nblk) {
  const int c   = blockIdx.x;
  const int tid = threadIdx.x;
  float s = 0.f, q = 0.f;
  for (int i = tid; i < nblk; i += 64) {
    s += ps[(size_t)c * nblk + i];
    q += pq[(size_t)c * nblk + i];
  }
  __shared__ float ss[64], sq[64];
  ss[tid] = s; sq[tid] = q;
  __syncthreads();
  for (int o = 32; o > 0; o >>= 1) {
    if (tid < o) { ss[tid] += ss[tid + o]; sq[tid] += sq[tid + o]; }
    __syncthreads();
  }
  if (tid == 0) {
    const float m   = ss[0] / (float)P;
    const float var = sq[0] / (float)P - m * m;
    const float sc  = g[c] * rsqrtf(var + 1e-5f);
    scale[c] = sc;
    shift[c] = be[c] - m * sc;
  }
}

__global__ void bn_stats_small_kernel(const float* __restrict__ in, const float* __restrict__ g,
                                      const float* __restrict__ be, float* __restrict__ scale,
                                      float* __restrict__ shift, int P, int C) {
  const int c   = blockIdx.x;
  const int tid = threadIdx.x;
  float s = 0.f, q = 0.f;
  for (int p = tid; p < P; p += 256) {
    const float v = in[(size_t)p * C + c];
    s += v;
    q += v * v;
  }
  __shared__ float ss[256], sq[256];
  ss[tid] = s; sq[tid] = q;
  __syncthreads();
  for (int o = 128; o > 0; o >>= 1) {
    if (tid < o) { ss[tid] += ss[tid + o]; sq[tid] += sq[tid + o]; }
    __syncthreads();
  }
  if (tid == 0) {
    const float m   = ss[0] / (float)P;
    const float var = sq[0] / (float)P - m * m;
    const float sc  = g[c] * rsqrtf(var + 1e-5f);
    scale[c] = sc;
    shift[c] = be[c] - m * sc;
  }
}

__global__ void bn_relu4_kernel(const float* __restrict__ in, const float* __restrict__ scale,
                                const float* __restrict__ shift, float* __restrict__ out,
                                int total4, int C) {
  const int idx = blockIdx.x * 256 + threadIdx.x;
  if (idx >= total4) return;
  const float4 v = ((const float4*)in)[idx];
  const int c0 = (idx * 4) % C;
  float4 r;
  r.x = v.x * scale[c0 + 0] + shift[c0 + 0];
  r.y = v.y * scale[c0 + 1] + shift[c0 + 1];
  r.z = v.z * scale[c0 + 2] + shift[c0 + 2];
  r.w = v.w * scale[c0 + 3] + shift[c0 + 3];
  r.x = r.x > 0.f ? r.x : 0.f;
  r.y = r.y > 0.f ? r.y : 0.f;
  r.z = r.z > 0.f ? r.z : 0.f;
  r.w = r.w > 0.f ? r.w : 0.f;
  ((float4*)out)[idx] = r;
}

__global__ void bn_sign_pack4_kernel(const float* __restrict__ in, const float* __restrict__ scale,
                                     const float* __restrict__ shift, int8_t* __restrict__ pk,
                                     int H, int W, int Hp, int Wp, int C, int Cp, int pad,
                                     int total4) {
  const int idx = blockIdx.x * 256 + threadIdx.x;
  if (idx >= total4) return;
  const int Cp4 = Cp >> 2;
  const int cq  = idx % Cp4;
  int t = idx / Cp4;
  const int wp = t % Wp; t /= Wp;
  const int hp = t % Hp;
  const int n  = t / Hp;
  const int h = hp - pad, w = wp - pad;
  int packed = 0;
  if (h >= 0 && h < H && w >= 0 && w < W) {
    const int c0 = cq * 4;
    const float* ip = in + ((size_t)(n * H + h) * W + w) * C;
    #pragma unroll
    for (int j = 0; j < 4; ++j) {
      const int c = c0 + j;
      int8_t v = 0;
      if (c < C) v = (ip[c] * scale[c] + shift[c]) >= 0.f ? (int8_t)1 : (int8_t)-1;
      packed |= ((int)(uint8_t)v) << (8 * j);
    }
  }
  ((int*)pk)[idx] = packed;
}

__global__ void pack_w_kernel(const float* __restrict__ w, int8_t* __restrict__ wp,
                              int O, int I, int K, int Cp, int total) {
  const int idx = blockIdx.x * 256 + threadIdx.x;
  if (idx >= total) return;
  const int ci = idx % Cp;
  int t = idx / Cp;
  const int kw = t % K; t /= K;
  const int kh = t % K;
  const int o  = t / K;
  int8_t v = 0;
  if (ci < I)
    v = w[((size_t)(o * I + ci) * K + kh) * K + kw] >= 0.f ? (int8_t)1 : (int8_t)-1;
  wp[idx] = v;
}

__global__ void __launch_bounds__(256)
binconv_iu8_kernel(const int8_t* __restrict__ act, const int8_t* __restrict__ wt,
                   const float* __restrict__ bias, float* __restrict__ out,
                   int HoWo, int Wo, int Hp, int Wp, int KH, int KW, int Cp, int Cout) {
  extern __shared__ int8_t smemA[];
  const int lane = threadIdx.x & 31;
  const int wave = threadIdx.x >> 5;
  const int half = (lane >> 4) & 1;
  const int col  = lane & 15;

  const int mtile  = blockIdx.y;
  const int ntile0 = (blockIdx.x * 8 + wave) * 4;
  const int Ktot   = KH * KW * Cp;

  {
    const int words = (16 * Ktot) >> 2;
    const int* gw = (const int*)(wt + (size_t)mtile * 16 * Ktot);
#if USE_ASYNC_LDS
    const uint32_t ldsbase = (uint32_t)(uintptr_t)smemA;
    for (int i = threadIdx.x; i < words; i += 256) {
      const int* gp = gw + i;
      const uint32_t lofs = ldsbase + (uint32_t)(i << 2);
      asm volatile("global_load_async_to_lds_b32 %0, %1, off"
                   :: "v"(lofs), "v"(gp) : "memory");
    }
#if __has_builtin(__builtin_amdgcn_s_wait_asynccnt)
    __builtin_amdgcn_s_wait_asynccnt(0);
#else
    asm volatile("s_wait_asynccnt 0x0" ::: "memory");
#endif
#else
    for (int i = threadIdx.x; i < words; i += 256)
      ((int*)smemA)[i] = gw[i];
#endif
    __syncthreads();
  }

  size_t base[4];
  int pix[4];
  #pragma unroll
  for (int t = 0; t < 4; ++t) {
    const int pixel = (ntile0 + t) * 16 + col;
    pix[t] = pixel;
    const int np = pixel / HoWo;
    const int pr = pixel - np * HoWo;
    const int oh = pr / Wo;
    const int ow = pr - oh * Wo;
    base[t] = (size_t)((np * Hp + oh) * Wp + ow) * Cp;
  }

  const int8_t* arow = smemA + col * Ktot;

  const v8i z = {0, 0, 0, 0, 0, 0, 0, 0};
  v8i acc[4] = {z, z, z, z};

  for (int kh = 0; kh < KH; ++kh) {
    for (int kw = 0; kw < KW; ++kw) {
      const int tap = (kh * Wp + kw) * Cp;
      const int8_t* aptr = arow + (kh * KW + kw) * Cp;
      for (int cb = 0; cb < Cp; cb += 64) {
        v8i a, b0, b1, b2, b3;
        #pragma unroll
        for (int v = 0; v < 8; ++v) {
          const int ka = ((v & 4) << 3) + ((v & 2) << 3) + ((v & 1) << 2) + (half << 3);
          a[v] = *(const int*)(aptr + cb + ka);
        }
        const int8_t* p0 = act + base[0] + tap + cb;
        const int8_t* p1 = act + base[1] + tap + cb;
        const int8_t* p2 = act + base[2] + tap + cb;
        const int8_t* p3 = act + base[3] + tap + cb;
        #pragma unroll
        for (int v = 0; v < 8; ++v) {
          const int kb = ((v & 4) << 3) + ((v & 2) << 3) + ((v & 1) << 2) + (half << 3);
          b0[v] = *(const int*)(p0 + kb);
          b1[v] = *(const int*)(p1 + kb);
          b2[v] = *(const int*)(p2 + kb);
          b3[v] = *(const int*)(p3 + kb);
        }
        acc[0] = __builtin_amdgcn_wmma_i32_16x16x64_iu8(true, a, true, b0, acc[0], false, false);
        acc[1] = __builtin_amdgcn_wmma_i32_16x16x64_iu8(true, a, true, b1, acc[1], false, false);
        acc[2] = __builtin_amdgcn_wmma_i32_16x16x64_iu8(true, a, true, b2, acc[2], false, false);
        acc[3] = __builtin_amdgcn_wmma_i32_16x16x64_iu8(true, a, true, b3, acc[3], false, false);
      }
    }
  }

  const int co0 = mtile * 16 + (half << 3);
  if (co0 < Cout) {
    float bb[8];
    #pragma unroll
    for (int r = 0; r < 8; ++r) bb[r] = bias[co0 + r];
    #pragma unroll
    for (int t = 0; t < 4; ++t) {
      float4 lo, hi;
      float vv;
      vv = (float)acc[t][0] + bb[0]; lo.x = vv > 0.f ? vv : 0.f;
      vv = (float)acc[t][1] + bb[1]; lo.y = vv > 0.f ? vv : 0.f;
      vv = (float)acc[t][2] + bb[2]; lo.z = vv > 0.f ? vv : 0.f;
      vv = (float)acc[t][3] + bb[3]; lo.w = vv > 0.f ? vv : 0.f;
      vv = (float)acc[t][4] + bb[4]; hi.x = vv > 0.f ? vv : 0.f;
      vv = (float)acc[t][5] + bb[5]; hi.y = vv > 0.f ? vv : 0.f;
      vv = (float)acc[t][6] + bb[6]; hi.z = vv > 0.f ? vv : 0.f;
      vv = (float)acc[t][7] + bb[7]; hi.w = vv > 0.f ? vv : 0.f;
      float* op = out + (size_t)pix[t] * Cout + co0;
      ((float4*)op)[0] = lo;
      ((float4*)op)[1] = hi;
    }
  }
}

__global__ void maxpool_kernel(const float* __restrict__ in, float* __restrict__ out,
                               int Ho, int Wo, int Hi, int Wi, int C, int total) {
  const int idx = blockIdx.x * 256 + threadIdx.x;
  if (idx >= total) return;
  const int c = idx % C;
  int t = idx / C;
  const int ow = t % Wo; t /= Wo;
  const int oh = t % Ho;
  const int n  = t / Ho;
  float m = -3.402823466e+38f;
  for (int i = 0; i < 3; ++i) {
    const int ih = oh * 2 - 1 + i;
    if (ih < 0 || ih >= Hi) continue;
    for (int j = 0; j < 3; ++j) {
      const int iw = ow * 2 - 1 + j;
      if (iw < 0 || iw >= Wi) continue;
      const float v = in[((size_t)(n * Hi + ih) * Wi + iw) * C + c];
      m = v > m ? v : m;
    }
  }
  out[idx] = m;
}

__global__ void avgpool_kernel(const float* __restrict__ in, float* __restrict__ out,
                               int Ho, int Wo, int Hi, int Wi, int C, int total) {
  const int idx = blockIdx.x * 256 + threadIdx.x;
  if (idx >= total) return;
  const int c = idx % C;
  int t = idx / C;
  const int ow = t % Wo; t /= Wo;
  const int oh = t % Ho;
  const int n  = t / Ho;
  float s = 0.f;
  for (int i = 0; i < 3; ++i) {
    const int ih = oh * 2 - 1 + i;
    if (ih < 0 || ih >= Hi) continue;
    for (int j = 0; j < 3; ++j) {
      const int iw = ow * 2 - 1 + j;
      if (iw < 0 || iw >= Wi) continue;
      s += in[((size_t)(n * Hi + ih) * Wi + iw) * C + c];
    }
  }
  out[idx] = s * (1.f / 9.f);
}

__global__ void conv8_kernel(const float* __restrict__ act, const float* __restrict__ w,
                             const float* __restrict__ bias, float* __restrict__ out, int total) {
  const int idx = blockIdx.x * 256 + threadIdx.x;
  if (idx >= total) return;
  const int co = idx % 10;
  const int p  = idx / 10;
  float s = bias[co];
  const float* ap = act + (size_t)p * 192;
  const float* wr = w + (size_t)co * 192;
  for (int ci = 0; ci < 192; ++ci) s += ap[ci] * wr[ci];
  out[idx] = s;
}

__global__ void final_mean_kernel(const float* __restrict__ c8, const float* __restrict__ scale,
                                  const float* __restrict__ shift, float* __restrict__ out,
                                  int total) {
  const int idx = blockIdx.x * 256 + threadIdx.x;
  if (idx >= total) return;
  const int c = idx % 10;
  const int n = idx / 10;
  float s = 0.f;
  for (int p = 0; p < 64; ++p) {
    const float v = c8[(size_t)(n * 64 + p) * 10 + c] * scale[c] + shift[c];
    s += v > 0.f ? v : 0.f;
  }
  out[idx] = s * (1.f / 64.f);
}

extern "C" void kernel_launch(void* const* d_in, const int* in_sizes, int n_in,
                              void* d_out, int out_size, void* d_ws, size_t ws_size,
                              hipStream_t stream) {
  const float* x = (const float*)d_in[0];
  const float *W[9], *CB[9], *G[9], *BE[9];
  for (int i = 0; i < 9; ++i) {
    W[i]  = (const float*)d_in[1 + 2 * i];
    CB[i] = (const float*)d_in[2 + 2 * i];
    G[i]  = (const float*)d_in[19 + 2 * i];
    BE[i] = (const float*)d_in[20 + 2 * i];
  }

  char* p = (char*)d_ws;
  auto take = [&](size_t bytes) -> char* {
    char* q = p;
    p += (bytes + 255) & ~(size_t)255;
    return q;
  };
  float*  bufA  = (float*)take((size_t)BATCH * 32 * 32 * 192 * 4);
  float*  bufB  = (float*)take((size_t)BATCH * 32 * 32 * 192 * 4);
  int8_t* pk    = (int8_t*)take((size_t)BATCH * 32 * 32 * 192);
  int8_t* wp    = (int8_t*)take(1u << 20);
  float*  scal  = (float*)take(192 * 4);
  float*  shif  = (float*)take(192 * 4);
  float*  c8    = (float*)take((size_t)BATCH * 64 * 10 * 4);
  float*  partS = (float*)take((size_t)192 * 256 * 4);
  float*  partQ = (float*)take((size_t)192 * 256 * 4);

  const int CHUNK = 2048;

  auto bn_stats = [&](const float* in, const float* g, const float* be, int P, int C) {
    const int nblk = (P + CHUNK - 1) / CHUNK;
    bn_part_kernel<<<nblk, C, 0, stream>>>(in, partS, partQ, P, C, CHUNK, nblk);
    bn_finish_kernel<<<C, 64, 0, stream>>>(partS, partQ, g, be, scal, shif, P, nblk);
  };

  float* cur = bufB;
  float* nxt = bufA;

  conv0_kernel<<<BATCH * 1024, 192, 0, stream>>>(x, W[0], CB[0], bufA);
  bn_stats(bufA, G[0], BE[0], BATCH * 1024, 192);
  {
    const int total4 = BATCH * 1024 * 192 / 4;
    bn_relu4_kernel<<<(total4 + 255) / 256, 256, 0, stream>>>(bufA, scal, shif, bufB,
                                                              total4, 192);
  }

  auto bin_layer = [&](int li, int H, int Wd, int Cin, int Cp, int Cout, int K, int pad) {
    const int P  = BATCH * H * Wd;
    const int Hp = H + 2 * pad, Wp2 = Wd + 2 * pad;
    bn_stats(cur, G[li], BE[li], P, Cin);
    const int pkTotal4 = BATCH * Hp * Wp2 * Cp / 4;
    bn_sign_pack4_kernel<<<(pkTotal4 + 255) / 256, 256, 0, stream>>>(
        cur, scal, shif, pk, H, Wd, Hp, Wp2, Cin, Cp, pad, pkTotal4);
    const int wTotal = Cout * K * K * Cp;
    pack_w_kernel<<<(wTotal + 255) / 256, 256, 0, stream>>>(W[li], wp, Cout, Cin, K, Cp, wTotal);
    dim3 grid(P / 512, Cout / 16);
    const int ldsBytes = 16 * K * K * Cp;
    binconv_iu8_kernel<<<grid, 256, ldsBytes, stream>>>(pk, wp, CB[li], nxt,
                                                        H * Wd, Wd, Hp, Wp2, K, K, Cp, Cout);
    float* t = cur; cur = nxt; nxt = t;
  };

  bin_layer(1, 32, 32, 192, 192, 160, 1, 0);
  bin_layer(2, 32, 32, 160, 192, 96, 1, 0);
  {
    const int total = BATCH * 16 * 16 * 96;
    maxpool_kernel<<<(total + 255) / 256, 256, 0, stream>>>(cur, nxt, 16, 16, 32, 32, 96, total);
    float* t = cur; cur = nxt; nxt = t;
  }
  bin_layer(3, 16, 16, 96, 128, 192, 5, 2);
  bin_layer(4, 16, 16, 192, 192, 192, 1, 0);
  bin_layer(5, 16, 16, 192, 192, 192, 1, 0);
  {
    const int total = BATCH * 8 * 8 * 192;
    avgpool_kernel<<<(total + 255) / 256, 256, 0, stream>>>(cur, nxt, 8, 8, 16, 16, 192, total);
    float* t = cur; cur = nxt; nxt = t;
  }
  bin_layer(6, 8, 8, 192, 192, 192, 3, 1);
  bin_layer(7, 8, 8, 192, 192, 192, 1, 0);

  {
    const int total = BATCH * 64 * 10;
    conv8_kernel<<<(total + 255) / 256, 256, 0, stream>>>(cur, W[8], CB[8], c8, total);
  }
  bn_stats_small_kernel<<<10, 256, 0, stream>>>(c8, G[8], BE[8], scal, shif, BATCH * 64, 10);
  {
    const int total = BATCH * 10;
    final_mean_kernel<<<(total + 255) / 256, 256, 0, stream>>>(c8, scal, shif,
                                                               (float*)d_out, total);
  }
}
